// AngularPenaltySMLoss_24223615550485
// MI455X (gfx1250) — compile-verified, matrix operand order checked by reading, not run on hardware
//
#include <hip/hip_runtime.h>
#include <hip/hip_bf16.h>

// ---------------------------------------------------------------------------
// ArcFace (AngularPenaltySMLoss) for MI455X / gfx1250, wave32 + WMMA + TDM.
//
// Pipeline (all on `stream`):
//   1) prep_x : L2-normalize rows of x, emit bf16 hi/lo split pair; zero rowsum
//   2) prep_w : split W into bf16 hi/lo pair
//   3) gemm   : wf = xn @ W.T via v_wmma_f32_16x16x32_bf16 (3 WMMAs / K=32 for
//               fp32-class accuracy). B panel staged to LDS once per block by
//               the Tensor Data Mover (removes 8x redundant L2 reads), fused
//               per-row sum(exp(64*wf)) epilogue with native f32 atomics.
//   4) loss   : gather label logit, ArcFace margin, mean, write d_out[0]
// ---------------------------------------------------------------------------

typedef __attribute__((ext_vector_type(16))) __bf16 v16bf;
typedef __attribute__((ext_vector_type(8)))  __bf16 bf16x8;
typedef __attribute__((ext_vector_type(8)))  float  v8f;
typedef __attribute__((ext_vector_type(4)))  unsigned int u32x4;
typedef __attribute__((ext_vector_type(4)))  int i32x4;
typedef __attribute__((ext_vector_type(8)))  int i32x8;

#define N_ROWS 4096
#define K_DIM  512
#define N_COLS 20000
#define S_SCALE 64.0f
#define EPS_CLIP 1e-7f
// cos(0.5), sin(0.5)
#define COS_M 0.87758256189037276f
#define SIN_M 0.47942553860420301f

// LDS B-panel: column stride 520 bf16 (512 data + 8 pad) -> 1040 B = 260
// DWORDs; 260 % 64 == 4, so lanes 0..15 hit banks {4m..4m+3}: conflict-free.
#define B_LDS_STRIDE 520
#define B_LDS_TENSOR (32 * B_LDS_STRIDE)   // elements per (hi|lo) tensor

// -------------------------------- helpers ----------------------------------

static __device__ __forceinline__ v16bf load16(const __bf16* p) {
  // A/B 16x32 bf16 tile, wave32: per lane 8 contiguous bf16 at +0 and 8 at +16
  bf16x8 lo = *(const bf16x8*)p;
  bf16x8 hi = *(const bf16x8*)(p + 16);
  return __builtin_shufflevector(lo, hi, 0, 1, 2, 3, 4, 5, 6, 7,
                                         8, 9, 10, 11, 12, 13, 14, 15);
}

static __device__ __forceinline__ v8f wmma_bf16(v16bf a, v16bf b, v8f c) {
  return __builtin_amdgcn_wmma_f32_16x16x32_bf16(
      /*neg_a=*/false, a, /*neg_b=*/false, b,
      /*c_mod=*/(short)0, c, /*reuse_a=*/false, /*reuse_b=*/false);
}

// Tensor Data Mover: DMA a 2D panel [32 cols x 512 K] of bf16 from global into
// LDS, inserting 16 B of padding after every 1024 B (one column) so the LDS
// image has a 520-element column stride.  D# packed per CDNA5 ISA 8.3-8.5.
static __device__ __forceinline__ void tdm_load_b_panel(const void* gsrc,
                                                        unsigned lds_byte_addr) {
  unsigned long long ga = (unsigned long long)gsrc;
  u32x4 g0;
  g0.x = 1u;                      // [1:0] count=1 (valid user descriptor)
  g0.y = lds_byte_addr;           // [63:32] lds_addr
  g0.z = (unsigned)ga;            // [95:64] global_addr[31:0]
  g0.w = (unsigned)((ga >> 32) & 0x01FFFFFFull) | (2u << 30); // addr[56:32]|type=2
  i32x8 g1;
  // w0: data_size=1 (2B) | pad_enable | pad_interval=7 (256 DW) | pad_amount=3 (4 DW)
  g1[0] = (1 << 16) | (1 << 20) | (7 << 22) | (3 << 25);
  g1[1] = (int)(512u << 16);      // [79:48] tensor_dim0 = 512 (low 16 bits)
  g1[2] = (int)(32u  << 16);      // [111:80] tensor_dim1 = 32 (low 16 bits)
  g1[3] = (int)(512u << 16);      // [127:112] tile_dim0 = 512
  g1[4] = 32;                     // [143:128] tile_dim1 = 32; tile_dim2 = 0
  g1[5] = 512;                    // [191:160] tensor_dim0_stride = 512
  g1[6] = 0;                      // dim0_stride hi / dim1_stride lo
  g1[7] = 0;
  i32x4 z4 = {0, 0, 0, 0};        // groups 2/3 unused (2D tensor)
#if defined(__clang_major__) && __clang_major__ >= 23
  i32x8 z8 = {0, 0, 0, 0, 0, 0, 0, 0};
  __builtin_amdgcn_tensor_load_to_lds(g0, g1, z4, z4, z8, 0);
#else
  __builtin_amdgcn_tensor_load_to_lds(g0, g1, z4, z4, 0);
#endif
}

static __device__ __forceinline__ unsigned lds_off(const void* p) {
  // Low 32 bits of a flat shared-aperture address are the LDS byte offset.
  return (unsigned)(unsigned long long)p;
}

// ------------------------------ kernel 1 -----------------------------------
// One wave per row: L2-normalize, split into bf16 hi/lo; lane0 zeroes rowsum.
__global__ __launch_bounds__(256) void prep_x_kernel(
    const float* __restrict__ x, __bf16* __restrict__ xh,
    __bf16* __restrict__ xl, float* __restrict__ rowsum) {
  const int lane = threadIdx.x & 31;
  const int wave = threadIdx.x >> 5;
  const int row  = blockIdx.x * 8 + wave;
  const float* xr = x + (size_t)row * K_DIM;

  float vals[16];
  float ss = 0.f;
#pragma unroll
  for (int j = 0; j < 16; ++j) {
    float v = xr[lane + 32 * j];
    vals[j] = v;
    ss += v * v;
  }
#pragma unroll
  for (int off = 1; off < 32; off <<= 1) ss += __shfl_xor(ss, off, 32);
  const float inv = 1.0f / fmaxf(sqrtf(ss), 1e-12f);

#pragma unroll
  for (int j = 0; j < 16; ++j) {
    float f = vals[j] * inv;
    __bf16 h = (__bf16)f;
    __bf16 l = (__bf16)(f - (float)h);
    xh[(size_t)row * K_DIM + lane + 32 * j] = h;
    xl[(size_t)row * K_DIM + lane + 32 * j] = l;
  }
  if (lane == 0) rowsum[row] = 0.0f;
}

// ------------------------------ kernel 2 -----------------------------------
__global__ __launch_bounds__(256) void prep_w_kernel(
    const float* __restrict__ W, __bf16* __restrict__ wh,
    __bf16* __restrict__ wl, int n) {
  int idx = blockIdx.x * blockDim.x + threadIdx.x;
  const int stride = gridDim.x * blockDim.x;
  for (; idx < n; idx += stride) {
    float f = W[idx];
    __bf16 h = (__bf16)f;
    wh[idx] = h;
    wl[idx] = (__bf16)(f - (float)h);
  }
}

// ------------------------------ kernel 3 -----------------------------------
// Block = 256 threads = 8 waves. Block tile: 128 rows x 32 cols.
// B panel (hi+lo) staged to LDS once per block via TDM; each wave computes
// one 16-row tile x two 16-col tiles over K=512 in steps of 32.
// Split-bf16 product: hi*hi + hi*lo + lo*hi  (lo*lo negligible).
__global__ __launch_bounds__(256) void gemm_arcface_kernel(
    const __bf16* __restrict__ Ah, const __bf16* __restrict__ Al,
    const __bf16* __restrict__ Bh, const __bf16* __restrict__ Bl,
    float* __restrict__ wf, float* __restrict__ rowsum) {
  __shared__ __bf16 smemB[2 * B_LDS_TENSOR];  // [hi | lo], 66,560 bytes

  const int lane    = threadIdx.x & 31;
  const int wave    = threadIdx.x >> 5;
  const int rowTile = blockIdx.y * 128 + wave * 16;
  const int col0    = blockIdx.x * 32;
  const int m       = lane & 15;        // A row / B col / C col within tile
  const int ksel    = (lane >> 4) << 3; // K sub-offset: 0 or 8

  if (wave == 0) {
    tdm_load_b_panel(Bh + (size_t)col0 * K_DIM, lds_off(&smemB[0]));
    tdm_load_b_panel(Bl + (size_t)col0 * K_DIM, lds_off(&smemB[B_LDS_TENSOR]));
    __builtin_amdgcn_s_wait_tensorcnt(0);
  }
  __syncthreads();

  const __bf16* arh = Ah + (size_t)(rowTile + m) * K_DIM + ksel;
  const __bf16* arl = Al + (size_t)(rowTile + m) * K_DIM + ksel;
  const __bf16* lb0h = &smemB[m * B_LDS_STRIDE + ksel];
  const __bf16* lb1h = &smemB[(m + 16) * B_LDS_STRIDE + ksel];
  const __bf16* lb0l = lb0h + B_LDS_TENSOR;
  const __bf16* lb1l = lb1h + B_LDS_TENSOR;

  v8f acc0 = {};
  v8f acc1 = {};

  for (int kk = 0; kk < K_DIM; kk += 32) {
    if (kk + 32 < K_DIM) {
      __builtin_prefetch(arh + kk + 32, 0, 1);
      __builtin_prefetch(arl + kk + 32, 0, 1);
    }
    v16bf a_h = load16(arh + kk);
    v16bf a_l = load16(arl + kk);
    v16bf p0h = load16(lb0h + kk);
    v16bf p0l = load16(lb0l + kk);
    v16bf p1h = load16(lb1h + kk);
    v16bf p1l = load16(lb1l + kk);

    acc0 = wmma_bf16(a_h, p0h, acc0);
    acc0 = wmma_bf16(a_h, p0l, acc0);
    acc0 = wmma_bf16(a_l, p0h, acc0);

    acc1 = wmma_bf16(a_h, p1h, acc1);
    acc1 = wmma_bf16(a_h, p1l, acc1);
    acc1 = wmma_bf16(a_l, p1h, acc1);
  }

  // Epilogue: store wf and fold exp(S*wf) into per-row sums.
  // C layout: VGPR v, lanes 0-15 -> (M=v, N=lane); lanes 16-31 -> (M=v+8).
  const int hb = (lane >> 4) << 3;
  float esum[8];
#pragma unroll
  for (int v = 0; v < 8; ++v) {
    const int r = rowTile + hb + v;
    wf[(size_t)r * N_COLS + col0 + m]      = acc0[v];
    wf[(size_t)r * N_COLS + col0 + 16 + m] = acc1[v];
    esum[v] = __expf(S_SCALE * acc0[v]) + __expf(S_SCALE * acc1[v]);
  }
  // Reduce across the 16 lanes of each half (xor offsets stay in-half).
#pragma unroll
  for (int v = 0; v < 8; ++v) {
#pragma unroll
    for (int off = 1; off < 16; off <<= 1)
      esum[v] += __shfl_xor(esum[v], off, 32);
  }
  if (m == 0) {
#pragma unroll
    for (int v = 0; v < 8; ++v)
      unsafeAtomicAdd(rowsum + rowTile + hb + v, esum[v]);
  }
}

// ------------------------------ kernel 4 -----------------------------------
__global__ __launch_bounds__(256) void loss_kernel(
    const float* __restrict__ wf, const int* __restrict__ labels,
    const float* __restrict__ rowsum, float* __restrict__ out) {
  __shared__ float red[8];
  float lsum = 0.f;
  for (int i = threadIdx.x; i < N_ROWS; i += 256) {
    const int lab = labels[i];
    const float d  = wf[(size_t)i * N_COLS + lab];
    const float dc = fminf(fmaxf(d, -1.f + EPS_CLIP), 1.f - EPS_CLIP);
    // s * cos(acos(dc) + m) == s * (dc*cos(m) - sqrt(1-dc^2)*sin(m))
    const float num   = S_SCALE * (dc * COS_M - sqrtf(fmaxf(1.f - dc * dc, 0.f)) * SIN_M);
    const float excl  = rowsum[i] - __expf(S_SCALE * d);
    const float denom = __expf(num) + excl;
    lsum += num - __logf(denom);
  }
#pragma unroll
  for (int off = 1; off < 32; off <<= 1) lsum += __shfl_xor(lsum, off, 32);
  const int lane = threadIdx.x & 31, wave = threadIdx.x >> 5;
  if (lane == 0) red[wave] = lsum;
  __syncthreads();
  if (threadIdx.x == 0) {
    float t = 0.f;
#pragma unroll
    for (int w = 0; w < 8; ++w) t += red[w];
    out[0] = -t / (float)N_ROWS;
  }
}

// ------------------------------ launcher -----------------------------------
extern "C" void kernel_launch(void* const* d_in, const int* in_sizes, int n_in,
                              void* d_out, int out_size, void* d_ws, size_t ws_size,
                              hipStream_t stream) {
  (void)in_sizes; (void)n_in; (void)out_size; (void)ws_size;
  const float* x      = (const float*)d_in[0];
  const int*   labels = (const int*)d_in[1];
  const float* W      = (const float*)d_in[2];

  float* out = (float*)d_out;
  float* wf  = out + 1;  // outputs: [loss, wf(4096x20000)] flat

  // Workspace layout (bytes):
  //   xn_hi: 4096*512*2  = 4,194,304      @ 0
  //   xn_lo: 4,194,304                    @ 4,194,304
  //   w_hi : 20000*512*2 = 20,480,000     @ 8,388,608
  //   w_lo : 20,480,000                   @ 28,868,608
  //   rowsum: 4096*4     = 16,384         @ 49,348,608
  char* ws = (char*)d_ws;
  __bf16* xh = (__bf16*)(ws + 0);
  __bf16* xl = (__bf16*)(ws + 4194304);
  __bf16* wh = (__bf16*)(ws + 8388608);
  __bf16* wl = (__bf16*)(ws + 28868608);
  float* rowsum = (float*)(ws + 49348608);

  prep_x_kernel<<<N_ROWS / 8, 256, 0, stream>>>(x, xh, xl, rowsum);
  prep_w_kernel<<<2048, 256, 0, stream>>>(W, wh, wl, N_COLS * K_DIM);

  dim3 grid(N_COLS / 32, N_ROWS / 128);
  gemm_arcface_kernel<<<grid, 256, 0, stream>>>(xh, xl, wh, wl, wf, rowsum);

  loss_kernel<<<1, 256, 0, stream>>>(wf, labels, rowsum, out);
}
